// MultiHeadAttention_43138651521394
// MI455X (gfx1250) — matrix operand order repaired and checked by reading, not compiled, not run on hardware
//
#include <hip/hip_runtime.h>
#include <math.h>

typedef __attribute__((ext_vector_type(16))) __bf16 v16bf;
typedef __attribute__((ext_vector_type(4)))  __bf16 v4bf;
typedef __attribute__((ext_vector_type(8)))  float  v8f;

constexpr int Bc = 4;      // batch
constexpr int Sc = 2048;   // sequence
constexpr int Dm = 1024;   // d_model == d_forward
constexpr int Hc = 16;     // heads
constexpr int Dk = 64;     // head dim
constexpr int Bk = 32;     // key block for flash attention

// ---------------------------------------------------------------------------
// WMMA helper (gfx1250, wave32). D = A(16x32 bf16) * B(32x16 bf16) + C(f32)
// ---------------------------------------------------------------------------
static __device__ __forceinline__ v8f wmma_bf16(v16bf a, v16bf b, v8f c) {
  return __builtin_amdgcn_wmma_f32_16x16x32_bf16(false, a, false, b,
                                                 (short)0, c, false, false);
}

// ---------------------------------------------------------------------------
// CDNA5 async global->LDS copy (ASYNCcnt path), 16 bytes per lane.
// LDS offset = low 32 bits of the flat shared-memory address (aperture rule).
// ---------------------------------------------------------------------------
static __device__ __forceinline__ unsigned lds_off(const void* p) {
  return (unsigned)(unsigned long long)p;
}
static __device__ __forceinline__ void async_load_b128(unsigned dst_lds,
                                                       const void* gsrc) {
  asm volatile("global_load_async_to_lds_b128 %0, %1, off"
               :: "v"(dst_lds), "v"(gsrc)
               : "memory");
}
static __device__ __forceinline__ void wait_async0() {
  asm volatile("s_wait_asynccnt 0" ::: "memory");
}
static __device__ __forceinline__ void wait_async2() {
  asm volatile("s_wait_asynccnt 2" ::: "memory");
}
static __device__ __forceinline__ void wait_async3() {
  asm volatile("s_wait_asynccnt 3" ::: "memory");
}

// A-matrix 16x32 bf16 fragment from a row-major [m][k] tile (lda in elements).
static __device__ __forceinline__ v16bf load_frag_a(const __bf16* tile, int lda) {
  int lane = threadIdx.x & 31;
  int m = lane & 15, half = lane >> 4;
  const __bf16* row = tile + (size_t)m * lda + half * 8;
  v16bf a;
#pragma unroll
  for (int i = 0; i < 8; ++i) {
    int kb = ((i & 4) << 2) + ((i & 3) << 1);
    a[2 * i]     = row[kb];
    a[2 * i + 1] = row[kb + 1];
  }
  return a;
}

// B-matrix 32x16 bf16 fragment from an [n][k] row-major tile (ldb in elements).
static __device__ __forceinline__ v16bf load_frag_b(const __bf16* tileT, int ldb) {
  int lane = threadIdx.x & 31;
  int n = lane & 15, half = lane >> 4;
  const __bf16* row = tileT + (size_t)n * ldb + half * 8;
  v16bf b;
#pragma unroll
  for (int j = 0; j < 16; ++j) b[j] = row[(j & 7) + ((j >> 3) << 4)];
  return b;
}

// ---------------------------------------------------------------------------
// fp32 -> bf16 conversion, 4 elements per thread (n must be multiple of 4)
// ---------------------------------------------------------------------------
__global__ void cvt_f32_bf16(const float* __restrict__ src,
                             __bf16* __restrict__ dst, int n4) {
  int i = blockIdx.x * blockDim.x + threadIdx.x;
  int stride = gridDim.x * blockDim.x;
  for (; i < n4; i += stride) {
    float4 f = ((const float4*)src)[i];
    v4bf o;
    o.x = (__bf16)f.x; o.y = (__bf16)f.y;
    o.z = (__bf16)f.z; o.w = (__bf16)f.w;
    ((v4bf*)dst)[i] = o;
  }
}

// ---------------------------------------------------------------------------
// QKV projection: O[b,h,s,n] = sum_m x[b,s,m] * W[h,n,m]
// KIND: 0 -> Q, 1 -> K (row-major [s][d]), 2 -> V transposed ([d][s]).
// grid = (S/128, B*H), block = 256 (8 waves). Triple-buffered async pipeline.
// ---------------------------------------------------------------------------
template <int KIND>
__global__ __launch_bounds__(256) void proj_kernel(
    const __bf16* __restrict__ X, const __bf16* __restrict__ Wall,
    __bf16* __restrict__ O) {
  __shared__ __align__(16) __bf16 ldsA[3][128 * 32];
  __shared__ __align__(16) __bf16 ldsB[3][64 * 32];

  int tid = threadIdx.x;
  int wave = tid >> 5, lane = tid & 31;
  int half = lane >> 4, ln = lane & 15;
  int bh = blockIdx.y;
  int b = bh >> 4, h = bh & 15;
  int s0 = blockIdx.x * 128;

  const __bf16* xrow = X + ((size_t)b * Sc + s0) * Dm;
  const __bf16* w = Wall + ((size_t)KIND * Hc + h) * (size_t)Dk * Dm;

  int arow = tid >> 1, ac = tid & 1;    // A: 128 rows x 32, 32B/thread
  int brow = tid >> 2, bc = tid & 3;    // B: 64 rows x 32, 16B/thread

  auto stage = [&](int t, int buf) {    // 3 async b128 per thread
    int m0 = t * 32;
    async_load_b128(lds_off(&ldsA[buf][arow * 32 + ac * 16]),
                    xrow + (size_t)arow * Dm + m0 + ac * 16);
    async_load_b128(lds_off(&ldsA[buf][arow * 32 + ac * 16 + 8]),
                    xrow + (size_t)arow * Dm + m0 + ac * 16 + 8);
    async_load_b128(lds_off(&ldsB[buf][brow * 32 + bc * 8]),
                    w + (size_t)brow * Dm + m0 + bc * 8);
  };

  v8f acc[4];
#pragma unroll
  for (int f = 0; f < 4; ++f)
#pragma unroll
    for (int r = 0; r < 8; ++r) acc[f][r] = 0.0f;

  constexpr int nT = Dm / 32;
  stage(0, 0);
  stage(1, 1);
  for (int t = 0; t < nT; ++t) {
    int buf = t % 3;
    if (t + 1 < nT) wait_async3(); else wait_async0();  // own tile-t loads done
    __syncthreads();                  // all waves' tile-t done; compute(t-1) done
    if (t + 2 < nT) stage(t + 2, (t + 2) % 3);

    v16bf a = load_frag_a(&ldsA[buf][wave * 16 * 32], 32);
#pragma unroll
    for (int f = 0; f < 4; ++f) {
      v16bf bb = load_frag_b(&ldsB[buf][f * 16 * 32], 32);
      acc[f] = wmma_bf16(a, bb, acc[f]);
    }
  }

  size_t base_bh = (size_t)bh * Sc * Dk;
#pragma unroll
  for (int f = 0; f < 4; ++f)
#pragma unroll
    for (int r = 0; r < 8; ++r) {
      int srow = s0 + wave * 16 + r + 8 * half;
      int n = f * 16 + ln;
      __bf16 val = (__bf16)acc[f][r];
      if constexpr (KIND == 2)
        O[(size_t)bh * Dk * Sc + (size_t)n * Sc + srow] = val;
      else
        O[base_bh + (size_t)srow * Dk + n] = val;
    }
}

// ---------------------------------------------------------------------------
// Flash attention per (b,h): scores never hit HBM. Each wave = 16 query rows,
// online softmax with the reference's zero-mask and *sqrt(d_k) scale.
// grid = (S/128, B*H), block = 256. Triple-buffered async K/V pipeline.
// ---------------------------------------------------------------------------
__global__ __launch_bounds__(256) void attn_kernel(
    const __bf16* __restrict__ Q, const __bf16* __restrict__ K,
    const __bf16* __restrict__ VT, __bf16* __restrict__ Z) {
  __shared__ __align__(16) __bf16 ldsK[3][Bk * Dk];   // [key][d]   3 x 4 KB
  __shared__ __align__(16) __bf16 ldsV[3][Dk * Bk];   // [d][key]   3 x 4 KB
  __shared__ __align__(16) __bf16 ldsP[8 * 16 * Bk];  // wave-local 8 KB

  int tid = threadIdx.x;
  int wave = tid >> 5, lane = tid & 31;
  int half = lane >> 4, ln = lane & 15;
  int bh = blockIdx.y;
  int b = bh >> 4, h = bh & 15;
  int qs0 = blockIdx.x * 128 + wave * 16;

  // Q fragments for this wave's 16 rows (held in regs for the whole kernel)
  const __bf16* qbase = Q + ((size_t)bh * Sc + qs0) * Dk;
  v16bf qa0 = load_frag_a(qbase, Dk);        // d = 0..31
  v16bf qa1 = load_frag_a(qbase + 32, Dk);   // d = 32..63

  const __bf16* kglob  = K  + (size_t)bh * Sc * Dk;
  const __bf16* vtglob = VT + (size_t)bh * Dk * Sc;

  int vrow = tid >> 2, vc = tid & 3;  // V^T staging: 64 rows x 32, 16B/thread

  auto stage = [&](int t, int buf) {  // 2 async b128 per thread
    async_load_b128(lds_off(&ldsK[buf][tid * 8]),
                    kglob + (size_t)t * Bk * Dk + tid * 8);
    async_load_b128(lds_off(&ldsV[buf][vrow * Bk + vc * 8]),
                    vtglob + (size_t)vrow * Sc + t * Bk + vc * 8);
  };

  v8f acc[4];
  float Mrow[8], Lrow[8];
#pragma unroll
  for (int f = 0; f < 4; ++f)
#pragma unroll
    for (int r = 0; r < 8; ++r) acc[f][r] = 0.0f;
#pragma unroll
  for (int r = 0; r < 8; ++r) { Mrow[r] = -3.0e38f; Lrow[r] = 0.0f; }

  constexpr int nT = Sc / Bk;  // 64 key blocks
  stage(0, 0);
  stage(1, 1);
  for (int t = 0; t < nT; ++t) {
    int buf = t % 3;
    if (t + 1 < nT) wait_async2(); else wait_async0();
    __syncthreads();
    if (t + 2 < nT) stage(t + 2, (t + 2) % 3);

    // ---- scores: 16 x 32 tile = 2 C fragments --------------------------
    v8f sc[2];
#pragma unroll
    for (int nb = 0; nb < 2; ++nb) {
#pragma unroll
      for (int r = 0; r < 8; ++r) sc[nb][r] = 0.0f;
      v16bf b0 = load_frag_b(&ldsK[buf][nb * 16 * Dk], Dk);       // d 0..31
      v16bf b1 = load_frag_b(&ldsK[buf][nb * 16 * Dk + 32], Dk);  // d 32..63
      sc[nb] = wmma_bf16(qa0, b0, sc[nb]);
      sc[nb] = wmma_bf16(qa1, b1, sc[nb]);
    }

    // ---- mask + scale + online softmax ---------------------------------
#pragma unroll
    for (int r = 0; r < 8; ++r) {
      float v0 = sc[0][r], v1 = sc[1][r];
      v0 = (v0 != 0.0f) ? v0 * 8.0f : -8.0e9f;  // where(s!=0,s,-1e9)*sqrt(64)
      v1 = (v1 != 0.0f) ? v1 * 8.0f : -8.0e9f;
      float mx = fmaxf(v0, v1);
#pragma unroll
      for (int off = 1; off < 16; off <<= 1)
        mx = fmaxf(mx, __shfl_xor(mx, off, 32));  // stays within lane-half
      float mnew = fmaxf(Mrow[r], mx);
      float alpha = __expf(Mrow[r] - mnew);
      Mrow[r] = mnew;
      float e0 = __expf(v0 - mnew), e1 = __expf(v1 - mnew);
      float rs = e0 + e1;
#pragma unroll
      for (int off = 1; off < 16; off <<= 1) rs += __shfl_xor(rs, off, 32);
      Lrow[r] = Lrow[r] * alpha + rs;
#pragma unroll
      for (int f = 0; f < 4; ++f) acc[f][r] *= alpha;
      sc[0][r] = e0;
      sc[1][r] = e1;
    }

    // ---- P: C-layout -> A-layout via wave-private LDS ------------------
    __bf16* pW = ldsP + wave * 16 * Bk;
#pragma unroll
    for (int r = 0; r < 8; ++r) {
      int m = r + 8 * half;
      pW[m * Bk + ln]      = (__bf16)sc[0][r];
      pW[m * Bk + 16 + ln] = (__bf16)sc[1][r];
    }
    asm volatile("s_wait_dscnt 0" ::: "memory");  // wave-local RAW on LDS

    // ---- Z += P(16x32) x V(32x64) --------------------------------------
    v16bf pa = load_frag_a(pW, Bk);
#pragma unroll
    for (int f = 0; f < 4; ++f) {
      v16bf vb = load_frag_b(&ldsV[buf][f * 16 * Bk], Bk);
      acc[f] = wmma_bf16(pa, vb, acc[f]);
    }
  }

  // epilogue: normalize and store Z bf16 at [b][s][h*64+d]
#pragma unroll
  for (int f = 0; f < 4; ++f)
#pragma unroll
    for (int r = 0; r < 8; ++r) {
      int srow = qs0 + r + 8 * half;
      int d = f * 16 + ln;
      float val = acc[f][r] / Lrow[r];
      Z[((size_t)b * Sc + srow) * Dm + h * 64 + d] = (__bf16)val;
    }
}

// ---------------------------------------------------------------------------
// Output projection: Out[s,o] = sum_f Z[s,f] * Wo[o,f] + bo[o]  (fp32 out)
// grid = (B*S/128, Dm/64), block = 256. Triple-buffered async pipeline.
// ---------------------------------------------------------------------------
__global__ __launch_bounds__(256) void oproj_kernel(
    const __bf16* __restrict__ Zb, const __bf16* __restrict__ Wob,
    const float* __restrict__ bo, float* __restrict__ Out) {
  __shared__ __align__(16) __bf16 ldsA[3][128 * 32];
  __shared__ __align__(16) __bf16 ldsB[3][64 * 32];

  int tid = threadIdx.x;
  int wave = tid >> 5, lane = tid & 31;
  int half = lane >> 4, ln = lane & 15;
  int s0 = blockIdx.x * 128;   // over B*S rows
  int n0 = blockIdx.y * 64;

  int arow = tid >> 1, ac = tid & 1;
  int brow = tid >> 2, bc = tid & 3;

  auto stage = [&](int t, int buf) {
    int m0 = t * 32;
    async_load_b128(lds_off(&ldsA[buf][arow * 32 + ac * 16]),
                    Zb + ((size_t)(s0 + arow)) * Dm + m0 + ac * 16);
    async_load_b128(lds_off(&ldsA[buf][arow * 32 + ac * 16 + 8]),
                    Zb + ((size_t)(s0 + arow)) * Dm + m0 + ac * 16 + 8);
    async_load_b128(lds_off(&ldsB[buf][brow * 32 + bc * 8]),
                    Wob + ((size_t)(n0 + brow)) * Dm + m0 + bc * 8);
  };

  v8f acc[4];
#pragma unroll
  for (int f = 0; f < 4; ++f)
#pragma unroll
    for (int r = 0; r < 8; ++r) acc[f][r] = 0.0f;

  constexpr int nT = Dm / 32;
  stage(0, 0);
  stage(1, 1);
  for (int t = 0; t < nT; ++t) {
    int buf = t % 3;
    if (t + 1 < nT) wait_async3(); else wait_async0();
    __syncthreads();
    if (t + 2 < nT) stage(t + 2, (t + 2) % 3);

    v16bf a = load_frag_a(&ldsA[buf][wave * 16 * 32], 32);
#pragma unroll
    for (int f = 0; f < 4; ++f) {
      v16bf bb = load_frag_b(&ldsB[buf][f * 16 * 32], 32);
      acc[f] = wmma_bf16(a, bb, acc[f]);
    }
  }

  float bias[4];
#pragma unroll
  for (int f = 0; f < 4; ++f) bias[f] = bo[n0 + f * 16 + ln];

#pragma unroll
  for (int f = 0; f < 4; ++f)
#pragma unroll
    for (int r = 0; r < 8; ++r) {
      int srow = s0 + wave * 16 + r + 8 * half;
      int o = n0 + f * 16 + ln;
      Out[(size_t)srow * Dm + o] = acc[f][r] + bias[f];
    }
}

// ---------------------------------------------------------------------------
extern "C" void kernel_launch(void* const* d_in, const int* in_sizes, int n_in,
                              void* d_out, int out_size, void* d_ws,
                              size_t ws_size, hipStream_t stream) {
  (void)in_sizes; (void)n_in; (void)out_size; (void)ws_size;
  const float* x  = (const float*)d_in[0];
  const float* Wq = (const float*)d_in[1];
  const float* Wk = (const float*)d_in[2];
  const float* Wv = (const float*)d_in[3];
  const float* Wo = (const float*)d_in[4];
  const float* bo = (const float*)d_in[5];
  float* out = (float*)d_out;

  // workspace layout (bf16 elements), total 88 MiB
  const size_t nX = (size_t)Bc * Sc * Dm;       // 8 Mi
  const size_t nW = (size_t)Hc * Dk * Dm;       // 1 Mi per tensor
  const size_t nP = (size_t)Bc * Hc * Sc * Dk;  // 8 Mi per q/k/v
  __bf16* xb   = (__bf16*)d_ws;
  __bf16* wall = xb + nX;               // [3][H][Dk][Dm]
  __bf16* wob  = wall + 3 * nW;         // [Dm][Dm]
  __bf16* qb   = wob + (size_t)Dm * Dm;
  __bf16* kb   = qb + nP;
  __bf16* vtb  = kb + nP;               // [b,h][d][s]
  __bf16* zb   = vtb + nP;              // [b][s][h*64+d]

  cvt_f32_bf16<<<1024, 256, 0, stream>>>(x, xb, (int)(nX / 4));
  cvt_f32_bf16<<<512, 256, 0, stream>>>(Wq, wall, (int)(nW / 4));
  cvt_f32_bf16<<<512, 256, 0, stream>>>(Wk, wall + nW, (int)(nW / 4));
  cvt_f32_bf16<<<512, 256, 0, stream>>>(Wv, wall + 2 * nW, (int)(nW / 4));
  cvt_f32_bf16<<<512, 256, 0, stream>>>(Wo, wob, Dm * Dm / 4);

  proj_kernel<0><<<dim3(Sc / 128, Bc * Hc), 256, 0, stream>>>(xb, wall, qb);
  proj_kernel<1><<<dim3(Sc / 128, Bc * Hc), 256, 0, stream>>>(xb, wall, kb);
  proj_kernel<2><<<dim3(Sc / 128, Bc * Hc), 256, 0, stream>>>(xb, wall, vtb);
  attn_kernel<<<dim3(Sc / 128, Bc * Hc), 256, 0, stream>>>(qb, kb, vtb, zb);
  oproj_kernel<<<dim3(Bc * Sc / 128, Dm / 64), 256, 0, stream>>>(zb, wob, bo, out);
}
